// PointNetSetAbstraction_69329362092243
// MI455X (gfx1250) — compile-verified, matrix operand order checked by reading, not run on hardware
//
#include <hip/hip_runtime.h>
#include <hip/hip_bf16.h>

typedef __attribute__((ext_vector_type(16))) _Float16 v16h;
typedef __attribute__((ext_vector_type(8)))  _Float16 v8h;
typedef __attribute__((ext_vector_type(8)))  float    v8f;
typedef __attribute__((ext_vector_type(4)))  float    f4;

#define NPTS   65536
#define MPTS   16384
#define NS     32
#define CIN    64
#define EPSBN  1e-5f

#define NWAVES 8
#define MPW    4
#define BLOCK  (NWAVES * 32)

// ---- LDS layout (bytes) ----
// wB  : 36 B-fragments * 32 lanes * 16 f16  = 18432 f16 = 36864 B
//       frag index: L0 -> kc*4+nt (12), L1 -> 12 + kc*4+nt (8), L2 -> 20 + kc*8+nt (16)
// sS  : 256 f32 scale   (ch 0..63 L0, 64..127 L1, 128..255 L2)
// sT  : 256 f32 bias
// act : per-wave 32 rows x 72 f16 (stride padded for bank-conflict-free frag loads)
#define WB_HALFS   18432
#define SS_OFF     36864
#define ST_OFF     (SS_OFF + 1024)
#define ACT_OFF    (ST_OFF + 1024)
#define ACT_STRIDE 72
#define ACT_WAVE   (32 * ACT_STRIDE)
#define SMEM_BYTES (ACT_OFF + NWAVES * ACT_WAVE * 2)   // 75776

static __device__ __forceinline__ v16h zero16() {
  v16h z;
#pragma unroll
  for (int i = 0; i < 16; ++i) z[i] = (_Float16)0.0f;
  return z;
}

// A-fragment element order for 16x16x32 f16 WMMA (per lane, khalf = lane>>4):
//   e0..7  -> K = khalf*8 + (0..7)
//   e8..15 -> K = 16 + khalf*8 + (0..7)
static __device__ __forceinline__ v16h pack16(f4 a, f4 b, f4 c, f4 d) {
  v16h r;
#pragma unroll
  for (int i = 0; i < 4; ++i) {
    r[i]      = (_Float16)a[i];
    r[4 + i]  = (_Float16)b[i];
    r[8 + i]  = (_Float16)c[i];
    r[12 + i] = (_Float16)d[i];
  }
  return r;
}

static __device__ __forceinline__ v16h joinh(v8h lo, v8h hi) {
  return __builtin_shufflevector(lo, hi, 0, 1, 2, 3, 4, 5, 6, 7,
                                 8, 9, 10, 11, 12, 13, 14, 15);
}

static __device__ __forceinline__ v8f wmma_f16(v16h a, v16h b, v8f c) {
  return __builtin_amdgcn_wmma_f32_16x16x32_f16(false, a, false, b,
                                                (short)0, c, false, false);
}

__global__ __launch_bounds__(BLOCK) void
pointnet_sa_kernel(const float* __restrict__ xyz,
                   const float* __restrict__ points,
                   const int* __restrict__ fps_idx,
                   const int* __restrict__ group_idx,
                   const float* __restrict__ w0, const float* __restrict__ b0,
                   const float* __restrict__ g0, const float* __restrict__ be0,
                   const float* __restrict__ m0, const float* __restrict__ v0,
                   const float* __restrict__ w1, const float* __restrict__ b1,
                   const float* __restrict__ g1, const float* __restrict__ be1,
                   const float* __restrict__ m1, const float* __restrict__ v1,
                   const float* __restrict__ w2, const float* __restrict__ b2,
                   const float* __restrict__ g2, const float* __restrict__ be2,
                   const float* __restrict__ m2, const float* __restrict__ v2,
                   float* __restrict__ out) {
  extern __shared__ char smem[];
  _Float16* wB  = (_Float16*)smem;
  float*    sS  = (float*)(smem + SS_OFF);
  float*    sT  = (float*)(smem + ST_OFF);
  _Float16* act = (_Float16*)(smem + ACT_OFF);

  const int tid   = threadIdx.x;
  const int wave  = tid >> 5;
  const int lane  = tid & 31;
  const int lhalf = lane >> 4;   // khalf
  const int l16   = lane & 15;
  _Float16* actw  = act + wave * ACT_WAVE;

  const int mblk = blockIdx.x * (NWAVES * MPW);   // 32 m per block

  // ---- new_xyz output (independent of LDS) ----
  if (tid < NWAVES * MPW * 3) {
    int ml = tid / 3, k = tid - ml * 3;
    int mi = mblk + ml;
    out[(size_t)mi * 3 + k] = xyz[(size_t)fps_idx[mi] * 3 + k];
  }
  if (blockIdx.x == 0 && tid == 0)
    ((int*)out)[(size_t)MPTS * 3 + (size_t)MPTS * 128] = MPTS;   // new_offset

  // ---- stage weights as pre-swizzled f16 B-fragments ----
  for (int i = tid; i < WB_HALFS; i += BLOCK) {
    int e  = i & 15;
    int ln = (i >> 4) & 31;
    int f  = i >> 9;
    int lh = ln >> 4, n16 = ln & 15;
    int kin = ((e & 8) << 1) | (lh << 3) | (e & 7);   // K within 32-chunk
    float val;
    if (f < 12) {                       // layer0: K=96 (pad), channel remap
      int kc = f >> 2, nt = f & 3;
      int k = kc * 32 + kin, n = nt * 16 + n16;
      int col = (k < 64) ? (k + 3) : ((k < 67) ? (k - 64) : -1);
      val = (col >= 0) ? w0[n * 67 + col] : 0.0f;
    } else if (f < 20) {                // layer1
      int gg = f - 12; int kc = gg >> 2, nt = gg & 3;
      val = w1[(nt * 16 + n16) * 64 + kc * 32 + kin];
    } else {                            // layer2
      int gg = f - 20; int kc = gg >> 3, nt = gg & 7;
      val = w2[(nt * 16 + n16) * 64 + kc * 32 + kin];
    }
    wB[i] = (_Float16)val;
  }

  // ---- fold conv bias + BN into per-channel scale/bias ----
  {
    int t = tid;
    if (t < 256) {
      const float *gp, *bp, *bep, *mp, *vp; int o;
      if (t < 64)       { o = t;       gp = g0; bp = b0; bep = be0; mp = m0; vp = v0; }
      else if (t < 128) { o = t - 64;  gp = g1; bp = b1; bep = be1; mp = m1; vp = v1; }
      else              { o = t - 128; gp = g2; bp = b2; bep = be2; mp = m2; vp = v2; }
      float s = gp[o] * rsqrtf(vp[o] + EPSBN);
      sS[t] = s;
      sT[t] = (bp[o] - mp[o]) * s + bep[o];
    }
  }
  __syncthreads();

  for (int it = 0; it < MPW; ++it) {
    const int m = mblk + wave * MPW + it;

    // ---- gather: neighbor indices + center ----
    int gidx[2];
    gidx[0] = group_idx[m * NS + l16];
    gidx[1] = group_idx[m * NS + 16 + l16];
    const int fi = fps_idx[m];
    const float nx = xyz[(size_t)fi * 3 + 0];
    const float ny = xyz[(size_t)fi * 3 + 1];
    const float nz = xyz[(size_t)fi * 3 + 2];

    // ---- build layer-0 A fragments straight from global (coalesced 256B rows) ----
    v16h A[3][2];   // [kchunk][row-tile]
#pragma unroll
    for (int rt = 0; rt < 2; ++rt) {
      const f4* pb = (const f4*)(points + (size_t)gidx[rt] * CIN) + lhalf * 2;
#pragma unroll
      for (int kc = 0; kc < 2; ++kc) {
        f4 a = pb[kc * 8], b = pb[kc * 8 + 1], c = pb[kc * 8 + 4], d = pb[kc * 8 + 5];
        A[kc][rt] = pack16(a, b, c, d);
      }
      v16h z = zero16();          // chunk 2: [dx,dy,dz, zeros...]
      if (lhalf == 0) {
        const float* xp = xyz + (size_t)gidx[rt] * 3;
        z[0] = (_Float16)(xp[0] - nx);
        z[1] = (_Float16)(xp[1] - ny);
        z[2] = (_Float16)(xp[2] - nz);
      }
      A[2][rt] = z;
    }

    // ---- layer 0: 32x96(f16) x 96x64 -> 32x64, scale+bias+relu -> act LDS ----
#pragma unroll
    for (int nt = 0; nt < 4; ++nt) {
      v8f acc0 = {0, 0, 0, 0, 0, 0, 0, 0};
      v8f acc1 = {0, 0, 0, 0, 0, 0, 0, 0};
#pragma unroll
      for (int kc = 0; kc < 3; ++kc) {
        v16h bf = *(const v16h*)&wB[(((kc << 2) + nt) * 32 + lane) * 16];
        acc0 = wmma_f16(A[kc][0], bf, acc0);
        acc1 = wmma_f16(A[kc][1], bf, acc1);
      }
      const int ch = nt * 16 + l16;
      const float s = sS[ch], t = sT[ch];
#pragma unroll
      for (int r = 0; r < 8; ++r) {
        float x0 = fmaxf(fmaf(acc0[r], s, t), 0.0f);
        float x1 = fmaxf(fmaf(acc1[r], s, t), 0.0f);
        actw[(r + lhalf * 8) * ACT_STRIDE + ch]      = (_Float16)x0;
        actw[(16 + r + lhalf * 8) * ACT_STRIDE + ch] = (_Float16)x1;
      }
    }

    // ---- layer 1: 32x64 x 64x64 -> 32x64 (load A frags fully, then overwrite act) ----
    v16h A1[2][2];
#pragma unroll
    for (int kc = 0; kc < 2; ++kc)
#pragma unroll
      for (int rt = 0; rt < 2; ++rt) {
        const _Float16* p = actw + (rt * 16 + l16) * ACT_STRIDE + kc * 32 + lhalf * 8;
        A1[kc][rt] = joinh(*(const v8h*)p, *(const v8h*)(p + 16));
      }
#pragma unroll
    for (int nt = 0; nt < 4; ++nt) {
      v8f acc0 = {0, 0, 0, 0, 0, 0, 0, 0};
      v8f acc1 = {0, 0, 0, 0, 0, 0, 0, 0};
#pragma unroll
      for (int kc = 0; kc < 2; ++kc) {
        v16h bf = *(const v16h*)&wB[((12 + (kc << 2) + nt) * 32 + lane) * 16];
        acc0 = wmma_f16(A1[kc][0], bf, acc0);
        acc1 = wmma_f16(A1[kc][1], bf, acc1);
      }
      const int ch = nt * 16 + l16;
      const float s = sS[64 + ch], t = sT[64 + ch];
#pragma unroll
      for (int r = 0; r < 8; ++r) {
        float x0 = fmaxf(fmaf(acc0[r], s, t), 0.0f);
        float x1 = fmaxf(fmaf(acc1[r], s, t), 0.0f);
        actw[(r + lhalf * 8) * ACT_STRIDE + ch]      = (_Float16)x0;
        actw[(16 + r + lhalf * 8) * ACT_STRIDE + ch] = (_Float16)x1;
      }
    }

    // ---- layer 2: 32x64 x 64x128 -> 32x128, relu, max-pool over 32 rows ----
    v16h A2[2][2];
#pragma unroll
    for (int kc = 0; kc < 2; ++kc)
#pragma unroll
      for (int rt = 0; rt < 2; ++rt) {
        const _Float16* p = actw + (rt * 16 + l16) * ACT_STRIDE + kc * 32 + lhalf * 8;
        A2[kc][rt] = joinh(*(const v8h*)p, *(const v8h*)(p + 16));
      }
#pragma unroll
    for (int nt = 0; nt < 8; ++nt) {
      v8f acc0 = {0, 0, 0, 0, 0, 0, 0, 0};
      v8f acc1 = {0, 0, 0, 0, 0, 0, 0, 0};
#pragma unroll
      for (int kc = 0; kc < 2; ++kc) {
        v16h bf = *(const v16h*)&wB[((20 + (kc << 3) + nt) * 32 + lane) * 16];
        acc0 = wmma_f16(A2[kc][0], bf, acc0);
        acc1 = wmma_f16(A2[kc][1], bf, acc1);
      }
      const int ch = nt * 16 + l16;
      const float s = sS[128 + ch], t = sT[128 + ch];
      float mx = 0.0f;                       // fmaxf(0,·) doubles as ReLU
#pragma unroll
      for (int r = 0; r < 8; ++r) {
        mx = fmaxf(mx, fmaf(acc0[r], s, t));
        mx = fmaxf(mx, fmaf(acc1[r], s, t));
      }
      mx = fmaxf(mx, __shfl_xor(mx, 16));    // combine the two row halves
      if (lane < 16)
        out[(size_t)MPTS * 3 + (size_t)m * 128 + ch] = mx;
    }
  }
}

extern "C" void kernel_launch(void* const* d_in, const int* in_sizes, int n_in,
                              void* d_out, int out_size, void* d_ws, size_t ws_size,
                              hipStream_t stream) {
  (void)in_sizes; (void)n_in; (void)out_size; (void)d_ws; (void)ws_size;
  const float* xyz       = (const float*)d_in[0];
  const float* points    = (const float*)d_in[1];
  const int*   fps_idx   = (const int*)d_in[2];
  const int*   group_idx = (const int*)d_in[3];
  const float* w0 = (const float*)d_in[4];  const float* b0 = (const float*)d_in[5];
  const float* g0 = (const float*)d_in[6];  const float* be0 = (const float*)d_in[7];
  const float* m0 = (const float*)d_in[8];  const float* v0 = (const float*)d_in[9];
  const float* w1 = (const float*)d_in[10]; const float* b1 = (const float*)d_in[11];
  const float* g1 = (const float*)d_in[12]; const float* be1 = (const float*)d_in[13];
  const float* m1 = (const float*)d_in[14]; const float* v1 = (const float*)d_in[15];
  const float* w2 = (const float*)d_in[16]; const float* b2 = (const float*)d_in[17];
  const float* g2 = (const float*)d_in[18]; const float* be2 = (const float*)d_in[19];
  const float* m2 = (const float*)d_in[20]; const float* v2 = (const float*)d_in[21];

  dim3 grid(MPTS / (NWAVES * MPW));   // 512 blocks, exact cover
  pointnet_sa_kernel<<<grid, BLOCK, SMEM_BYTES, stream>>>(
      xyz, points, fps_idx, group_idx,
      w0, b0, g0, be0, m0, v0,
      w1, b1, g1, be1, m1, v1,
      w2, b2, g2, be2, m2, v2,
      (float*)d_out);
}
